// MultiHeadAtt_5196910428660
// MI455X (gfx1250) — matrix operand order repaired and checked by reading, not compiled, not run on hardware
//
#include <hip/hip_runtime.h>
#include <hip/hip_bf16.h>
#include <stdint.h>

// ---------------------------------------------------------------------------
// MultiHeadAttention (causal), bf16 WMMA pipeline for gfx1250 (MI455X).
// B=2, N=M=2048, D=2048, H=16, DK=128.
// ---------------------------------------------------------------------------

typedef __attribute__((ext_vector_type(16))) __bf16 v16bf;
typedef __attribute__((ext_vector_type(2)))  __bf16 v2bf;
typedef __attribute__((ext_vector_type(8)))  float  v8f;
typedef __attribute__((ext_vector_type(4)))  int    v4i;

#define GAS __attribute__((address_space(1)))
#define LAS __attribute__((address_space(3)))

union Frag {
  uint32_t u[8];
  v16bf    v;
};

constexpr int BB  = 2;
constexpr int NN  = 2048;
constexpr int DD  = 2048;
constexpr int HH  = 16;
constexpr int DK  = 128;

// ---- fp32 -> bf16 conversion helpers --------------------------------------
__device__ __forceinline__ uint32_t bfrnd(float f) {
  // fp32 bits + RNE increment toward bf16 (keep in high half)
  union { float f; uint32_t u; } x{f};
  return x.u + 0x7FFFu + ((x.u >> 16) & 1u);
}

__device__ __forceinline__ uint16_t f2bf(float f) {
  return (uint16_t)(bfrnd(f) >> 16);
}

__device__ __forceinline__ uint32_t pkbf(float lo, float hi) {
#if __has_builtin(__builtin_amdgcn_cvt_pk_bf16_f32)
  v2bf r = __builtin_amdgcn_cvt_pk_bf16_f32(lo, hi);
  return __builtin_bit_cast(uint32_t, r);
#else
  // one v_perm_b32 selects bytes {lo.b2, lo.b3, hi.b2, hi.b3}
  return __builtin_amdgcn_perm(bfrnd(hi), bfrnd(lo), 0x07060302u);
#endif
}

// ---- CDNA5 async global->LDS copy (ASYNCcnt-tracked) ----------------------
__device__ __forceinline__ void async_copy_b128(const uint16_t* g, uint16_t* l) {
#if __has_builtin(__builtin_amdgcn_global_load_async_to_lds_b128)
  __builtin_amdgcn_global_load_async_to_lds_b128((GAS v4i*)g, (LAS v4i*)l, 0, 0);
#else
  const uint32_t loff  = (uint32_t)(uintptr_t)(LAS void*)l;
  const uint64_t gaddr = (uint64_t)(uintptr_t)g;
  asm volatile("global_load_async_to_lds_b128 %0, %1, off"
               :: "v"(loff), "v"(gaddr) : "memory");
#endif
}

__device__ __forceinline__ void wait_async0() {
#if __has_builtin(__builtin_amdgcn_s_wait_asynccnt)
  __builtin_amdgcn_s_wait_asynccnt(0);
#else
  asm volatile("s_wait_asynccnt 0x0" ::: "memory");
#endif
}

__device__ __forceinline__ v8f wmma_bf16(v16bf a, v16bf b, v8f c) {
  // D = A(16x32 bf16) * B(32x16 bf16) + C(16x16 f32)
  return __builtin_amdgcn_wmma_f32_16x16x32_bf16(false, a, false, b, (short)0, c,
                                                 false, false);
}

// ---------------------------------------------------------------------------
// Kernel 1: Y(bf16, [B][H][S][DK]) = X(f32, [B*S][D]) @ W(f32,[D][D])^T
// Block: 256 threads (8 waves). Tile: 64 rows x 128 cols (one head).
// ---------------------------------------------------------------------------
__global__ __launch_bounds__(256)
void gemm_qkv_kernel(const float* __restrict__ X, const float* __restrict__ W,
                     uint16_t* __restrict__ Y) {
  __shared__ uint16_t Xs[64 * 32];
  __shared__ uint16_t Ws[128 * 32];

  const int t       = threadIdx.x;
  const int lane    = t & 31;
  const int wave    = t >> 5;
  const int laneRow = lane & 15;
  const int hi      = lane >> 4;           // 0: lanes 0-15, 1: lanes 16-31
  const int rowBase = blockIdx.x * 64;
  const int h       = blockIdx.y;
  const int colBase = h * DK;
  const int wr      = (wave >> 2) * 32;    // wave row offset in 64-row tile
  const int wc      = (wave & 3) * 32;     // wave col offset in 128-col tile

  const v8f vz = {0.f,0.f,0.f,0.f,0.f,0.f,0.f,0.f};
  v8f acc[2][2] = {{vz, vz}, {vz, vz}};

  const int kaoff = hi * 8;                // A-layout K offset per lane-half
  const int kboff = hi * 16;               // unused; B loads use kaoff too

  for (int k0 = 0; k0 < DD; k0 += 32) {
    __syncthreads();
    {   // stage X tile 64x32 (f32 -> bf16, packed dword stores)
      const int r = t >> 2, c = (t & 3) * 8;
      const float* src = X + (size_t)(rowBase + r) * DD + k0 + c;
      float4 a = *(const float4*)src;
      float4 b = *(const float4*)(src + 4);
      uint32_t* dst = (uint32_t*)&Xs[r * 32 + c];
      dst[0] = pkbf(a.x, a.y); dst[1] = pkbf(a.z, a.w);
      dst[2] = pkbf(b.x, b.y); dst[3] = pkbf(b.z, b.w);
      if (k0 + 32 < DD) __builtin_prefetch(src + 32, 0, 1);
    }
    {   // stage W tile 128x32 (f32 -> bf16, packed dword stores)
      const int r = t >> 1, c = (t & 1) * 16;
      const float* src = W + (size_t)(colBase + r) * DD + k0 + c;
      uint32_t* dst = (uint32_t*)&Ws[r * 32 + c];
      #pragma unroll
      for (int j = 0; j < 4; ++j) {
        float4 a = *(const float4*)(src + j * 4);
        dst[2*j]   = pkbf(a.x, a.y);
        dst[2*j+1] = pkbf(a.z, a.w);
      }
      if (k0 + 32 < DD) __builtin_prefetch(src + 32, 0, 1);
    }
    __syncthreads();

    Frag af[2], bfr[2];
    #pragma unroll
    for (int s = 0; s < 2; ++s) {
      const int arow = wr + s * 16 + laneRow;
      #pragma unroll
      for (int d = 0; d < 8; ++d) {        // A: k = e + (e>=8?8:0) + 8*hi
        const int k = (d < 4 ? 2*d : 2*d + 8) + kaoff;
        af[s].u[d] = *(const uint32_t*)&Xs[arow * 32 + k];
      }
      const int bcol = wc + s * 16 + laneRow;
      #pragma unroll
      for (int d = 0; d < 8; ++d) {        // B: k = e + (e>=8?8:0) + 8*hi
        const int k = (d < 4 ? 2*d : 2*d + 8) + kaoff;
        bfr[s].u[d] = *(const uint32_t*)&Ws[bcol * 32 + k];
      }
    }
    #pragma unroll
    for (int r = 0; r < 2; ++r)
      #pragma unroll
      for (int s = 0; s < 2; ++s)
        acc[r][s] = wmma_bf16(af[r].v, bfr[s].v, acc[r][s]);
  }

  // store: head-split bf16 layout [b][h][n][dk]
  #pragma unroll
  for (int r = 0; r < 2; ++r) {
    #pragma unroll
    for (int s = 0; s < 2; ++s) {
      const int dk = wc + s * 16 + laneRow;
      #pragma unroll
      for (int i = 0; i < 8; ++i) {
        const int gr = rowBase + wr + r * 16 + i + hi * 8;
        const int bb = gr >> 11;           // row / N
        const int n  = gr & (NN - 1);
        Y[(((size_t)bb * HH + h) * NN + n) * DK + dk] = f2bf(acc[r][s][i]);
      }
    }
  }
}

// ---------------------------------------------------------------------------
// Kernel 2: flash attention per (128-query block, head, batch).
// 8 waves/WG; wave w owns 16 query rows. K/V staged in LDS per 32-key step.
// K tile staged with CDNA5 async global->LDS; V staged with v_perm pair-packed
// transpose so B-fragment key pairs are contiguous dwords.
// ---------------------------------------------------------------------------
__global__ __launch_bounds__(256)
void attn_kernel(const uint16_t* __restrict__ Qb, const uint16_t* __restrict__ Kb,
                 const uint16_t* __restrict__ Vb, uint16_t* __restrict__ msg) {
  __shared__ uint16_t Ks[32 * 128];        // [m][dk]
  __shared__ uint16_t Vt[128 * 32];        // transposed: [dk][m]
  __shared__ uint16_t Ps[8][16 * 32];      // per-wave P staging (A-layout rows)

  const int t       = threadIdx.x;
  const int lane    = t & 31;
  const int wave    = t >> 5;
  const int laneRow = lane & 15;
  const int hi      = lane >> 4;
  const int blk     = blockIdx.x;
  const int h       = blockIdx.y;
  const int b       = blockIdx.z;

  const size_t headOff = ((size_t)b * HH + h) * NN * DK;
  const uint16_t* Qp = Qb + headOff;
  const uint16_t* Kp = Kb + headOff;
  const uint16_t* Vp = Vb + headOff;

  const int qBase = blk * 128 + wave * 16;
  const int kaoff = hi * 8;
  const int kboff = hi * 16;

  // Load Q fragments (16 rows x 128 dk => 4 A-fragments) straight from global.
  Frag qf[4];
  {
    const int n = qBase + laneRow;
    #pragma unroll
    for (int f = 0; f < 4; ++f)
      #pragma unroll
      for (int d = 0; d < 8; ++d) {
        const int k = f * 32 + (d < 4 ? 2*d : 2*d + 8) + kaoff;
        qf[f].u[d] = *(const uint32_t*)&Qp[(size_t)n * DK + k];
      }
  }

  const v8f vz = {0.f,0.f,0.f,0.f,0.f,0.f,0.f,0.f};
  v8f o[8] = {vz, vz, vz, vz, vz, vz, vz, vz};
  float mst[8], lst[8];
  #pragma unroll
  for (int i = 0; i < 8; ++i) { mst[i] = -1.0e30f; lst[i] = 0.0f; }

  const float scale = 0.08838834764831845f;   // 1/sqrt(128)
  const int lim = blk * 128 + 128;            // causal bound for this WG

  for (int kt = 0; kt < lim; kt += 32) {
    __syncthreads();
    {   // stage K tile 32x128 via async global->LDS (b128 per lane)
      const int r = t >> 3, c = (t & 7) * 16;
      const uint16_t* src = &Kp[(size_t)(kt + r) * DK + c];
      async_copy_b128(src,     &Ks[r * 128 + c]);
      async_copy_b128(src + 8, &Ks[r * 128 + c + 8]);
      if (kt + 32 < lim)
        __builtin_prefetch(&Kp[(size_t)(kt + 32 + r) * DK + c], 0, 1);
    }
    {   // stage V tile 32x128, v_perm pair-packed transpose into Vt[dk][m]
      const int mp = t & 15;            // m pair: rows 2*mp, 2*mp+1
      const int c  = (t >> 4) * 8;      // dk octet
      const uint32_t* r0 = (const uint32_t*)&Vp[(size_t)(kt + 2*mp)     * DK + c];
      const uint32_t* r1 = (const uint32_t*)&Vp[(size_t)(kt + 2*mp + 1) * DK + c];
      #pragma unroll
      for (int j = 0; j < 4; ++j) {
        const uint32_t w0 = r0[j], w1 = r1[j];
        uint32_t* d0 = (uint32_t*)&Vt[(c + 2*j)     * 32 + 2*mp];
        uint32_t* d1 = (uint32_t*)&Vt[(c + 2*j + 1) * 32 + 2*mp];
        *d0 = __builtin_amdgcn_perm(w1, w0, 0x05040100u);  // {V[m][dk],   V[m+1][dk]}
        *d1 = __builtin_amdgcn_perm(w1, w0, 0x07060302u);  // {V[m][dk+1], V[m+1][dk+1]}
      }
      if (kt + 32 < lim)
        __builtin_prefetch(&Vp[(size_t)(kt + 32 + 2*mp) * DK + c], 0, 1);
    }
    wait_async0();
    __syncthreads();

    // S = Q * K^T for two 16-key sub-tiles (C layout f32)
    v8f sT[2];
    #pragma unroll
    for (int e = 0; e < 2; ++e) {
      v8f s = vz;
      const int bcol = e * 16 + laneRow;   // key index in 32-key tile
      #pragma unroll
      for (int f = 0; f < 4; ++f) {
        Frag bk;
        #pragma unroll
        for (int d = 0; d < 8; ++d) {
          const int k = f * 32 + (d < 4 ? 2*d : 2*d + 8) + kaoff;
          bk.u[d] = *(const uint32_t*)&Ks[bcol * 128 + k];
        }
        s = wmma_bf16(qf[f].v, bk.v, s);
      }
      sT[e] = s;
    }

    // causal mask + scale (C layout: row = i + 8*hi, col = laneRow).
    // Fast path: tile entirely below the diagonal for all 16 rows of this
    // wave (wave-uniform branch) -> no compare/select needed.
    if (kt + 31 <= qBase) {
      #pragma unroll
      for (int e = 0; e < 2; ++e)
        #pragma unroll
        for (int i = 0; i < 8; ++i)
          sT[e][i] *= scale;
    } else {
      #pragma unroll
      for (int e = 0; e < 2; ++e) {
        const int mg = kt + e * 16 + laneRow;
        #pragma unroll
        for (int i = 0; i < 8; ++i) {
          const int q = qBase + i + hi * 8;
          const float v = sT[e][i] * scale;
          sT[e][i] = (mg <= q) ? v : -3.0e38f;
        }
      }
    }

    // online softmax update (row reductions across 16-lane halves)
    #pragma unroll
    for (int i = 0; i < 8; ++i) {
      float mx = fmaxf(sT[0][i], sT[1][i]);
      #pragma unroll
      for (int off = 1; off < 16; off <<= 1)
        mx = fmaxf(mx, __shfl_xor(mx, off, 32));
      mx = fmaxf(mx, mst[i]);
      const float alpha = __expf(mst[i] - mx);
      const float p0 = __expf(sT[0][i] - mx);
      const float p1 = __expf(sT[1][i] - mx);
      sT[0][i] = p0; sT[1][i] = p1;
      float rs = p0 + p1;
      #pragma unroll
      for (int off = 1; off < 16; off <<= 1)
        rs += __shfl_xor(rs, off, 32);
      lst[i] = lst[i] * alpha + rs;
      mst[i] = mx;
      #pragma unroll
      for (int f = 0; f < 8; ++f) o[f][i] *= alpha;
    }

    // P: C-layout -> LDS -> A-layout fragment (wave-local, DS ops in-order)
    #pragma unroll
    for (int e = 0; e < 2; ++e)
      #pragma unroll
      for (int i = 0; i < 8; ++i)
        Ps[wave][(i + hi * 8) * 32 + e * 16 + laneRow] = f2bf(sT[e][i]);

    Frag pf;
    #pragma unroll
    for (int d = 0; d < 8; ++d) {
      const int k = (d < 4 ? 2*d : 2*d + 8) + kaoff;
      pf.u[d] = *(const uint32_t*)&Ps[wave][laneRow * 32 + k];
    }

    // O += P * V  (8 dk tiles of 16)
    #pragma unroll
    for (int f = 0; f < 8; ++f) {
      Frag vf;
      const int vcol = f * 16 + laneRow;   // dk column
      #pragma unroll
      for (int d = 0; d < 8; ++d) {
        const int k = (d < 4 ? 2*d : 2*d + 8) + kaoff;         // key index
        vf.u[d] = *(const uint32_t*)&Vt[vcol * 32 + k];
      }
      o[f] = wmma_bf16(pf.v, vf.v, o[f]);
    }
  }

  // normalize and store msg (bf16, [b][n][d])
  #pragma unroll
  for (int i = 0; i < 8; ++i) {
    const float inv = 1.0f / lst[i];
    const int nrow = qBase + i + hi * 8;
    #pragma unroll
    for (int f = 0; f < 8; ++f) {
      msg[((size_t)(b * NN + nrow)) * DD + h * DK + f * 16 + laneRow] =
          f2bf(o[f][i] * inv);
    }
  }
}

// ---------------------------------------------------------------------------
// Kernel 3: out(f32) = msg(bf16) @ Wo(f32)^T + bo
// X tile staged via async global->LDS (pure bf16 copy).
// ---------------------------------------------------------------------------
__global__ __launch_bounds__(256)
void gemm_out_kernel(const uint16_t* __restrict__ Xb, const float* __restrict__ W,
                     const float* __restrict__ bias, float* __restrict__ out) {
  __shared__ uint16_t Xs[64 * 32];
  __shared__ uint16_t Ws[128 * 32];

  const int t       = threadIdx.x;
  const int lane    = t & 31;
  const int wave    = t >> 5;
  const int laneRow = lane & 15;
  const int hi      = lane >> 4;
  const int rowBase = blockIdx.x * 64;
  const int colBase = blockIdx.y * 128;
  const int wr      = (wave >> 2) * 32;
  const int wc      = (wave & 3) * 32;

  const v8f vz = {0.f,0.f,0.f,0.f,0.f,0.f,0.f,0.f};
  v8f acc[2][2] = {{vz, vz}, {vz, vz}};

  const int kaoff = hi * 8;
  const int kboff = hi * 16;

  for (int k0 = 0; k0 < DD; k0 += 32) {
    __syncthreads();
    {   // stage X tile 64x32 (bf16 straight copy, async b128)
      const int r = t >> 2, c = (t & 3) * 8;
      const uint16_t* src = &Xb[(size_t)(rowBase + r) * DD + k0 + c];
      async_copy_b128(src, &Xs[r * 32 + c]);
      if (k0 + 32 < DD) __builtin_prefetch(src + 32, 0, 1);
    }
    {   // stage W tile 128x32 (f32 -> bf16, packed dword stores)
      const int r = t >> 1, c = (t & 1) * 16;
      const float* src = W + (size_t)(colBase + r) * DD + k0 + c;
      uint32_t* dst = (uint32_t*)&Ws[r * 32 + c];
      #pragma unroll
      for (int j = 0; j < 4; ++j) {
        float4 a = *(const float4*)(src + j * 4);
        dst[2*j]   = pkbf(a.x, a.y);
        dst[2*j+1] = pkbf(a.z, a.w);
      }
      if (k0 + 32 < DD) __builtin_prefetch(src + 32, 0, 1);
    }
    wait_async0();
    __syncthreads();

    Frag af[2], bfr[2];
    #pragma unroll
    for (int s = 0; s < 2; ++s) {
      const int arow = wr + s * 16 + laneRow;
      #pragma unroll
      for (int d = 0; d < 8; ++d) {
        const int k = (d < 4 ? 2*d : 2*d + 8) + kaoff;
        af[s].u[d] = *(const uint32_t*)&Xs[arow * 32 + k];
      }
      const int bcol = wc + s * 16 + laneRow;
      #pragma unroll
      for (int d = 0; d < 8; ++d) {
        const int k = (d < 4 ? 2*d : 2*d + 8) + kaoff;
        bfr[s].u[d] = *(const uint32_t*)&Ws[bcol * 32 + k];
      }
    }
    #pragma unroll
    for (int r = 0; r < 2; ++r)
      #pragma unroll
      for (int s = 0; s < 2; ++s)
        acc[r][s] = wmma_bf16(af[r].v, bfr[s].v, acc[r][s]);
  }

  #pragma unroll
  for (int r = 0; r < 2; ++r) {
    #pragma unroll
    for (int s = 0; s < 2; ++s) {
      const int col = colBase + wc + s * 16 + laneRow;
      const float bv = bias[col];
      #pragma unroll
      for (int i = 0; i < 8; ++i) {
        const int gr = rowBase + wr + r * 16 + i + hi * 8;
        out[(size_t)gr * DD + col] = acc[r][s][i] + bv;
      }
    }
  }
}

// ---------------------------------------------------------------------------
extern "C" void kernel_launch(void* const* d_in, const int* in_sizes, int n_in,
                              void* d_out, int out_size, void* d_ws, size_t ws_size,
                              hipStream_t stream) {
  (void)in_sizes; (void)n_in; (void)out_size; (void)ws_size;

  const float* tgt    = (const float*)d_in[0];   // [B][N][D]
  const float* memory = (const float*)d_in[1];   // [B][M][D]
  const float* Wq     = (const float*)d_in[2];
  const float* Wk     = (const float*)d_in[3];
  const float* Wv     = (const float*)d_in[4];
  const float* Wo     = (const float*)d_in[5];
  const float* bo     = (const float*)d_in[6];
  // d_in[7] = tgt_mask: causal triu(k=1), implemented analytically in attn.

  const size_t headElems = (size_t)BB * HH * NN * DK;   // 8M bf16 elems each
  uint16_t* Qb  = (uint16_t*)d_ws;
  uint16_t* Kb  = Qb + headElems;
  uint16_t* Vb  = Kb + headElems;
  uint16_t* msg = Vb + headElems;                       // [B*N][D] bf16

  const dim3 gGemm((BB * NN) / 64, DD / 128);           // 64 x 16
  gemm_qkv_kernel<<<gGemm, 256, 0, stream>>>(tgt,    Wq, Qb);
  gemm_qkv_kernel<<<gGemm, 256, 0, stream>>>(memory, Wk, Kb);
  gemm_qkv_kernel<<<gGemm, 256, 0, stream>>>(memory, Wv, Vb);

  const dim3 gAttn(NN / 128, HH, BB);                   // 16 x 16 x 2
  attn_kernel<<<gAttn, 256, 0, stream>>>(Qb, Kb, Vb, msg);

  gemm_out_kernel<<<gGemm, 256, 0, stream>>>(msg, Wo, bo, (float*)d_out);
}
